// GNN9_19378892439969
// MI455X (gfx1250) — matrix operand order repaired and checked by reading, not compiled, not run on hardware
//
#include <hip/hip_runtime.h>
#include <hip/hip_bf16.h>
#include <math.h>

// ---------------------------------------------------------------------------
// Types for CDNA5 WMMA (wave32, 16x16x32 bf16 -> f32)
// ---------------------------------------------------------------------------
typedef __attribute__((ext_vector_type(16))) __bf16 v16bf;
typedef __attribute__((ext_vector_type(8)))  float  v8f;

__device__ __forceinline__ unsigned short f2bf(float f) {
  unsigned int u = __float_as_uint(f);
  u += 0x7FFFu + ((u >> 16) & 1u);   // round-to-nearest-even
  return (unsigned short)(u >> 16);
}

// ---------------------------------------------------------------------------
// Fill
// ---------------------------------------------------------------------------
__global__ void k_fill(float* __restrict__ p, float v, long long n) {
  long long t = (long long)blockIdx.x * blockDim.x + threadIdx.x;
  if (t < n) p[t] = v;
}

// ---------------------------------------------------------------------------
// Embedding gather: xb[n, 0:IN] = bf16(emb[tokens[n], 0:IN])
// ---------------------------------------------------------------------------
__global__ void k_embed(const int* __restrict__ tok, const float* __restrict__ emb,
                        unsigned short* __restrict__ xb, int N, int IN) {
  long long t = (long long)blockIdx.x * blockDim.x + threadIdx.x;
  long long total = (long long)N * (IN >> 2);
  if (t >= total) return;
  int n = (int)(t / (IN >> 2));
  int c = (int)(t % (IN >> 2)) << 2;
  float4 v = *(const float4*)(emb + (size_t)tok[n] * IN + c);
  unsigned short* p = xb + (size_t)n * IN + c;
  p[0] = f2bf(v.x); p[1] = f2bf(v.y); p[2] = f2bf(v.z); p[3] = f2bf(v.w);
}

// ---------------------------------------------------------------------------
// f32 -> bf16 array convert (n must be multiple of 4)
// ---------------------------------------------------------------------------
__global__ void k_f2bf(const float* __restrict__ in, unsigned short* __restrict__ out,
                       long long n4) {
  long long t = (long long)blockIdx.x * blockDim.x + threadIdx.x;
  if (t >= n4) return;
  float4 v = ((const float4*)in)[t];
  unsigned short* p = out + t * 4;
  p[0] = f2bf(v.x); p[1] = f2bf(v.y); p[2] = f2bf(v.z); p[3] = f2bf(v.w);
}

// ---------------------------------------------------------------------------
// Weight prep: W [K, Nc] f32  ->  Wt [Nc, K] bf16 (transposed)
// ---------------------------------------------------------------------------
__global__ void k_wtrans(const float* __restrict__ W, unsigned short* __restrict__ Wt,
                         int K, int Nc) {
  int t = blockIdx.x * blockDim.x + threadIdx.x;
  if (t >= K * Nc) return;
  int k = t / Nc, n = t % Nc;
  Wt[(size_t)n * K + k] = f2bf(W[t]);
}

// ---------------------------------------------------------------------------
// WMMA GEMM:  C[M, Nc] = A[M, K](bf16) @ W[K, Nc]
//   A row-major bf16, Wt = W^T stored [Nc, K] bf16, C f32 row-major.
//   One wave -> one 16x16 tile. Requires M%16==0, K%32==0, Nc%16==0.
// A-frag layout (16-bit 16x32): lane L: M=L&15, half=L>>4,
//   elems = K in {half*8 + 0..7} then {16 + half*8 + 0..7}  (two 16B loads)
// B-frag layout (32x16):        lane L: N=L&15, K = {half*8 + 0..7} then {16 + half*8 + 0..7} (two 16B loads
//   from the transposed-weight row)
// C/D layout: VGPR r -> row M = r + 8*half, col N = L&15.
// ---------------------------------------------------------------------------
__global__ void k_gemm_bf16(const unsigned short* __restrict__ A,
                            const unsigned short* __restrict__ Wt,
                            float* __restrict__ C,
                            int Mtiles, int K, int Nc) {
  int mt = blockIdx.x * blockDim.y + threadIdx.y;
  if (mt >= Mtiles) return;
  int nt   = blockIdx.y;
  int lane = threadIdx.x;
  int half = lane >> 4;
  int l16  = lane & 15;

  const unsigned short* ap = A  + (size_t)(mt * 16 + l16) * K;
  const unsigned short* bp = Wt + (size_t)(nt * 16 + l16) * K;

  v8f acc = {};
  for (int k0 = 0; k0 < K; k0 += 32) {
    union { v16bf v; uint4 u[2]; } a, b;
    a.u[0] = *(const uint4*)(ap + k0 + half * 8);
    a.u[1] = *(const uint4*)(ap + k0 + 16 + half * 8);
    b.u[0] = *(const uint4*)(bp + k0 + half * 8);
    b.u[1] = *(const uint4*)(bp + k0 + 16 + half * 8);
    // D = A*B + C   (8 args: neg_a, A, neg_b, B, c_mod, C, reuse_a, reuse_b)
    acc = __builtin_amdgcn_wmma_f32_16x16x32_bf16(false, a.v, false, b.v,
                                                  (short)0, acc, false, false);
  }
  float* cp = C + (size_t)(mt * 16 + half * 8) * Nc + nt * 16 + l16;
#pragma unroll
  for (int r = 0; r < 8; ++r) cp[(size_t)r * Nc] = acc[r];
}

// ---------------------------------------------------------------------------
// Float atomic max via CAS (handles any sign, init = -inf)
// ---------------------------------------------------------------------------
__device__ __forceinline__ void atomicMaxF(float* addr, float val) {
  int* ai = reinterpret_cast<int*>(addr);
  int cur = *ai;
  while (__int_as_float(cur) < val) {
    int prev = atomicCAS(ai, cur, __float_as_int(val));
    if (prev == cur) break;
    cur = prev;
  }
}

// ---------------------------------------------------------------------------
// Edge pass 1: alpha[e,h] = sum_c leaky_relu(xl[src,h,c]+xr[dst,h,c]+ee[e,h,c]) * att[h,c]
//              and segment-max into amax[dst,h]. One thread per (edge, head).
// ---------------------------------------------------------------------------
__global__ void k_edge_alpha(const float* __restrict__ xl, const float* __restrict__ xr,
                             const float* __restrict__ ee, const int* __restrict__ src,
                             const int* __restrict__ dst, const float* __restrict__ att,
                             float* __restrict__ alpha, float* __restrict__ amax,
                             int E, int C) {
  long long t = (long long)blockIdx.x * blockDim.x + threadIdx.x;
  if (t >= (long long)E * 2) return;
  int e = (int)(t >> 1), h = (int)(t & 1);
  int s = src[e], d = dst[e];
  int HC = 2 * C;
  const float4* pl = (const float4*)(xl + (size_t)s * HC + h * C);
  const float4* pr = (const float4*)(xr + (size_t)d * HC + h * C);
  const float4* pe = (const float4*)(ee + (size_t)e * HC + h * C);
  const float4* pa = (const float4*)(att + h * C);
  float sum = 0.f;
  for (int i = 0; i < C / 4; ++i) {
    float4 a = pl[i], b = pr[i], c = pe[i], w = pa[i];
    float m;
    m = a.x + b.x + c.x; m = (m > 0.f) ? m : 0.2f * m; sum += m * w.x;
    m = a.y + b.y + c.y; m = (m > 0.f) ? m : 0.2f * m; sum += m * w.y;
    m = a.z + b.z + c.z; m = (m > 0.f) ? m : 0.2f * m; sum += m * w.z;
    m = a.w + b.w + c.w; m = (m > 0.f) ? m : 0.2f * m; sum += m * w.w;
  }
  alpha[t] = sum;
  atomicMaxF(&amax[(size_t)d * 2 + h], sum);
}

// ---------------------------------------------------------------------------
// Edge pass 2: ex = exp(alpha - amax[dst]); den[dst] += ex  (alpha overwritten)
// ---------------------------------------------------------------------------
__global__ void k_edge_exp(float* __restrict__ alpha, const int* __restrict__ dst,
                           const float* __restrict__ amax, float* __restrict__ den, int E) {
  long long t = (long long)blockIdx.x * blockDim.x + threadIdx.x;
  if (t >= (long long)E * 2) return;
  int e = (int)(t >> 1), h = (int)(t & 1);
  int d = dst[e];
  float ex = __expf(alpha[t] - amax[(size_t)d * 2 + h]);
  alpha[t] = ex;
  atomicAdd(&den[(size_t)d * 2 + h], ex);
}

// ---------------------------------------------------------------------------
// Edge pass 3: acc[dst,h,c] += (ex/den[dst,h]) * xl[src,h,c]
//   32 lanes per edge, each lane does 4 consecutive channels of the 2C=128.
// ---------------------------------------------------------------------------
__global__ void k_edge_aggr(const float* __restrict__ ex, const float* __restrict__ den,
                            const float* __restrict__ xl, const int* __restrict__ src,
                            const int* __restrict__ dst, float* __restrict__ acc,
                            int E, int C) {
  long long t = (long long)blockIdx.x * blockDim.x + threadIdx.x;
  int e = (int)(t >> 5);
  if (e >= E) return;
  int lane = (int)(t & 31);
  int HC = 2 * C;
  int ch = lane * (HC / 32);      // HC=128 -> 4 channels/lane, head boundary aligned
  int h  = ch / C;
  int s = src[e], d = dst[e];
  float a = ex[(size_t)e * 2 + h] / den[(size_t)d * 2 + h];
  float4 v = *(const float4*)(xl + (size_t)s * HC + ch);
  float* pd = acc + (size_t)d * HC + ch;
  atomicAdd(pd + 0, a * v.x);
  atomicAdd(pd + 1, a * v.y);
  atomicAdd(pd + 2, a * v.z);
  atomicAdd(pd + 3, a * v.w);
}

// ---------------------------------------------------------------------------
// Finalize: mean over heads + bias (+ ReLU); emit bf16 activation and/or f32.
// ---------------------------------------------------------------------------
__global__ void k_finalize(const float* __restrict__ acc, const float* __restrict__ bias,
                           unsigned short* __restrict__ xb_next, float* __restrict__ xoutf,
                           int N, int C, int relu) {
  long long t = (long long)blockIdx.x * blockDim.x + threadIdx.x;
  if (t >= (long long)N * C) return;
  int n = (int)(t / C), c = (int)(t % C);
  int HC = 2 * C;
  float v = 0.5f * (acc[(size_t)n * HC + c] + acc[(size_t)n * HC + C + c]) + bias[c];
  if (relu) v = fmaxf(v, 0.f);
  if (xb_next) xb_next[t] = f2bf(v);
  if (xoutf)   xoutf[t]   = v;
}

// ---------------------------------------------------------------------------
// Master pooling: out[g,:] = x[lower_bound(batch, g), :]
// ---------------------------------------------------------------------------
__global__ void k_pool(const float* __restrict__ x, const int* __restrict__ batch,
                       float* __restrict__ out, int N, int C) {
  int g = blockIdx.x;
  __shared__ int master;
  if (threadIdx.x == 0) {
    int lo = 0, hi = N;
    while (lo < hi) { int mid = (lo + hi) >> 1; if (batch[mid] < g) lo = mid + 1; else hi = mid; }
    master = lo;
  }
  __syncthreads();
  out[(size_t)g * C + threadIdx.x] = x[(size_t)master * C + threadIdx.x];
}

// ---------------------------------------------------------------------------
// Host-side orchestration
// ---------------------------------------------------------------------------
static inline size_t g1(long long n, int b) { return (size_t)((n + b - 1) / b); }

extern "C" void kernel_launch(void* const* d_in, const int* in_sizes, int n_in,
                              void* d_out, int out_size, void* d_ws, size_t ws_size,
                              hipStream_t stream) {
  const int*   tokens    = (const int*)  d_in[0];
  const int*   edge_idx  = (const int*)  d_in[1];
  const float* edge_attr = (const float*)d_in[2];
  const int*   batch     = (const int*)  d_in[3];
  const float* emb       = (const float*)d_in[5];

  const int N  = in_sizes[0];
  const int E  = in_sizes[1] / 2;
  const int HC = in_sizes[9];          // H*C = 128
  const int C  = HC / 2;               // 64 (H = 2)
  const int IN = in_sizes[6] / HC;     // 512
  const int DE = in_sizes[8] / HC;     // 64
  const int MID = in_sizes[11] / HC;   // 64
  const int B  = out_size / C;         // num graphs

  const int* src = edge_idx;
  const int* dst = edge_idx + E;

  const int wlIdx[3] = {6, 11, 16}, wrIdx[3] = {7, 12, 17}, weIdx[3] = {8, 13, 18};
  const int atIdx[3] = {9, 14, 19}, bIdx[3]  = {10, 15, 20};
  const int Kl[3] = {IN, MID, MID};

  // ---- workspace carve-out (256B aligned slabs) ----
  char* ws = (char*)d_ws;
  size_t off = 0;
  auto take = [&](size_t bytes) -> void* {
    off = (off + 255) & ~(size_t)255;
    void* p = ws + off; off += bytes; return p;
  };
  unsigned short* xb  = (unsigned short*)take((size_t)N * IN * 2);
  unsigned short* eab = (unsigned short*)take((size_t)E * DE * 2);
  unsigned short *Wlt[3], *Wrt[3], *Wet[3];
  for (int l = 0; l < 3; ++l) {
    Wlt[l] = (unsigned short*)take((size_t)HC * Kl[l] * 2);
    Wrt[l] = (unsigned short*)take((size_t)HC * Kl[l] * 2);
    Wet[l] = (unsigned short*)take((size_t)HC * DE * 2);
  }
  float* xl   = (float*)take((size_t)N * HC * 4);
  float* xr   = (float*)take((size_t)N * HC * 4);
  float* ee   = (float*)take((size_t)E * HC * 4);
  float* alp  = (float*)take((size_t)E * 2 * 4);
  float* amax = (float*)take((size_t)N * 2 * 4);
  float* den  = (float*)take((size_t)N * 2 * 4);
  float* acc  = (float*)take((size_t)N * HC * 4);
  unsigned short* xb2 = (unsigned short*)take((size_t)N * C * 2);
  float* xout = (float*)take((size_t)N * C * 4);
  (void)ws_size; (void)n_in;

  // ---- one-time prep: embedding gather, edge_attr convert, weight transposes ----
  k_embed<<<g1((long long)N * (IN / 4), 256), 256, 0, stream>>>(tokens, emb, xb, N, IN);
  k_f2bf<<<g1((long long)E * DE / 4, 256), 256, 0, stream>>>(edge_attr, eab, (long long)E * DE / 4);
  for (int l = 0; l < 3; ++l) {
    k_wtrans<<<g1((long long)Kl[l] * HC, 256), 256, 0, stream>>>((const float*)d_in[wlIdx[l]], Wlt[l], Kl[l], HC);
    k_wtrans<<<g1((long long)Kl[l] * HC, 256), 256, 0, stream>>>((const float*)d_in[wrIdx[l]], Wrt[l], Kl[l], HC);
    k_wtrans<<<g1((long long)DE * HC, 256), 256, 0, stream>>>((const float*)d_in[weIdx[l]], Wet[l], DE, HC);
  }

  const dim3 gblk(32, 4);
  const int Mtn = N / 16, Mte = E / 16;

  for (int l = 0; l < 3; ++l) {
    const unsigned short* act = (l == 0) ? xb : xb2;
    const float* att  = (const float*)d_in[atIdx[l]];
    const float* bias = (const float*)d_in[bIdx[l]];

    // node and edge GEMMs (WMMA)
    k_gemm_bf16<<<dim3((Mtn + 3) / 4, HC / 16), gblk, 0, stream>>>(act, Wlt[l], xl, Mtn, Kl[l], HC);
    k_gemm_bf16<<<dim3((Mtn + 3) / 4, HC / 16), gblk, 0, stream>>>(act, Wrt[l], xr, Mtn, Kl[l], HC);
    k_gemm_bf16<<<dim3((Mte + 3) / 4, HC / 16), gblk, 0, stream>>>(eab, Wet[l], ee, Mte, DE, HC);

    // reset segment buffers
    k_fill<<<g1((long long)N * 2, 256), 256, 0, stream>>>(amax, -INFINITY, (long long)N * 2);
    k_fill<<<g1((long long)N * 2, 256), 256, 0, stream>>>(den, 0.f, (long long)N * 2);
    k_fill<<<g1((long long)N * HC, 256), 256, 0, stream>>>(acc, 0.f, (long long)N * HC);

    // attention + scatter softmax + aggregation
    k_edge_alpha<<<g1((long long)E * 2, 256), 256, 0, stream>>>(xl, xr, ee, src, dst, att, alp, amax, E, C);
    k_edge_exp<<<g1((long long)E * 2, 256), 256, 0, stream>>>(alp, dst, amax, den, E);
    k_edge_aggr<<<g1((long long)E * 32, 256), 256, 0, stream>>>(alp, den, xl, src, dst, acc, E, C);

    // head mean + bias (+ReLU), produce next activation / final features
    if (l < 2)
      k_finalize<<<g1((long long)N * C, 256), 256, 0, stream>>>(acc, bias, xb2, nullptr, N, C, 1);
    else
      k_finalize<<<g1((long long)N * C, 256), 256, 0, stream>>>(acc, bias, nullptr, xout, N, C, 0);
  }

  // master-node pooling -> d_out [B, C] f32
  k_pool<<<B, C, 0, stream>>>(xout, batch, (float*)d_out, N, C);
}
